// GNN_45878840656168
// MI455X (gfx1250) — matrix operand order repaired and checked by reading, not compiled, not run on hardware
//
#include <hip/hip_runtime.h>
#include <hip/hip_bf16.h>

// ---------------- CDNA5 (gfx1250, wave32) GNN implementation ----------------
// D = 128 fixed. bf16 WMMA (v_wmma_f32_16x16x32_bf16) for all dense GEMMs,
// fragment-layout LDS tiles so each WMMA operand is two ds_load_b128.
// Wave-per-edge kernels for the sparse attention phase.

#define DD 128
#define BSTR 136   // bf16 LDS tile stride: 272B row => bank stride 4, 16B aligned

typedef __attribute__((ext_vector_type(8)))  __bf16 v8bf;
typedef __attribute__((ext_vector_type(16))) __bf16 v16bf;
typedef __attribute__((ext_vector_type(8)))  float  v8f;

__device__ __forceinline__ v16bf cat8(v8bf lo, v8bf hi) {
    return __builtin_shufflevector(lo, hi, 0,1,2,3,4,5,6,7,8,9,10,11,12,13,14,15);
}

// ---------------------------------------------------------------------------
// Triple GEMM: Cj[M x 128] = A[M x 128] @ Bj[128 x 128] + biasj  (j = 0..2,
// null Bj skipped). f32 in/out, bf16 WMMA MACs. Block = 128 threads = 4 waves;
// each wave computes a 16-row x 128-col strip (8 WMMA tiles, K in 4 chunks).
// A tile staged once in bf16 (row-major); each B staged bf16 TRANSPOSED
// (column-major) so every B fragment is two contiguous 8-bf16 runs (2x ds_load_b128).
// ---------------------------------------------------------------------------
__global__ __launch_bounds__(128)
void gnn_gemm3(const float* __restrict__ A,
               const float* __restrict__ B0, const float* __restrict__ B1,
               const float* __restrict__ B2,
               const float* __restrict__ b0, const float* __restrict__ b1,
               const float* __restrict__ b2,
               float* __restrict__ C0, float* __restrict__ C1,
               float* __restrict__ C2, int M) {
    __shared__ __attribute__((aligned(16))) __bf16 As[64 * BSTR];   // [m][k]
    __shared__ __attribute__((aligned(16))) __bf16 Bs[DD * BSTR];   // [n][k] (transposed)
    __shared__ float biasS[DD];

    const int t    = threadIdx.x;      // 0..127
    const int wave = t >> 5;
    const int lane = t & 31;
    const int half = lane >> 4;        // 0/1
    const int l16  = lane & 15;
    const int row0 = blockIdx.x * 64;

    const float* Bp[3] = {B0, B1, B2};
    const float* bp[3] = {b0, b1, b2};
    float*       Cp[3] = {C0, C1, C2};

    // Stage A tile (64 rows) as bf16, coalesced, zero-filled past M.
    for (int i = 0; i < 64; ++i) {
        int row = row0 + i;
        As[i * BSTR + t] = (__bf16)((row < M) ? A[(size_t)row * DD + t] : 0.0f);
    }
    __syncthreads();

    // Preload the 4 A fragments for this wave's 16 rows.
    // A frag element e: K = kc*32 + (e>>3)*16 + half*8 + (e&7)
    //  -> two contiguous 8-elem runs at [kc*32 + half*8] and [+16].
    const int mloc = wave * 16 + l16;
    v16bf afrag[4];
#pragma unroll
    for (int kc = 0; kc < 4; ++kc) {
        const __bf16* ab = &As[mloc * BSTR + kc * 32 + half * 8];
        afrag[kc] = cat8(*(const v8bf*)ab, *(const v8bf*)(ab + 16));
    }

    for (int j = 0; j < 3; ++j) {
        if (!Bp[j]) continue;                 // uniform across block
        __syncthreads();                      // prior iteration done with Bs
        const float* B = Bp[j];
        // Stage B transposed: thread t owns column t; read coalesced rows.
        for (int k = 0; k < DD; ++k)
            Bs[t * BSTR + k] = (__bf16)B[k * DD + t];
        biasS[t] = bp[j] ? bp[j][t] : 0.0f;
        __syncthreads();

        v8f acc[8];
#pragma unroll
        for (int n = 0; n < 8; ++n) acc[n] = (v8f){0.f,0.f,0.f,0.f,0.f,0.f,0.f,0.f};

#pragma unroll
        for (int kc = 0; kc < 4; ++kc) {
#pragma unroll
            for (int n = 0; n < 8; ++n) {
                // B frag element e: K = kc*32 + (e>>3)*16 + half*8 + (e&7), col = n*16 + l16
                //  -> two contiguous 8-elem runs in the transposed tile.
                const __bf16* bb = &Bs[(n * 16 + l16) * BSTR + kc * 32 + half * 8];
                v16bf bfrag = cat8(*(const v8bf*)bb, *(const v8bf*)(bb + 16));
                acc[n] = __builtin_amdgcn_wmma_f32_16x16x32_bf16(
                    false, afrag[kc], false, bfrag, (short)0, acc[n], false, false);
            }
        }

        // C/D layout: VGPR r -> M = r + 8*half (within wave tile), N = l16.
        float* C = Cp[j];
#pragma unroll
        for (int n = 0; n < 8; ++n) {
            int col = n * 16 + l16;
            float bv = biasS[col];
#pragma unroll
            for (int r = 0; r < 8; ++r) {
                int grow = row0 + wave * 16 + r + half * 8;
                if (grow < M) C[(size_t)grow * DD + col] = acc[n][r] + bv;
            }
        }
    }
}

// bias_fused[n] = sum_d b[d] * eW[d][n]   (128 threads, one block)
__global__ void gnn_bias_fuse(const float* __restrict__ b, const float* __restrict__ eW,
                              float* __restrict__ bf) {
    int n = threadIdx.x;
    float acc = 0.f;
    for (int d = 0; d < DD; ++d) acc += b[d] * eW[d * DD + n];
    bf[n] = acc;
}

// ---------------------------------------------------------------------------
// Edge phase. Order-preserving float<->uint for atomicMax-based segment max.
// ---------------------------------------------------------------------------
__device__ __forceinline__ unsigned f2ord(float f) {
    unsigned b = __float_as_uint(f);
    return (b & 0x80000000u) ? ~b : (b | 0x80000000u);
}
__device__ __forceinline__ float ord2f(unsigned u) {
    unsigned b = (u & 0x80000000u) ? (u & 0x7FFFFFFFu) : ~u;
    return __uint_as_float(b);
}

// One wave32 per edge: s = leaky_relu(dot(qW[src], k[dst]) / sqrt(128))
__global__ void gnn_edge_score(const float* __restrict__ qW, const float* __restrict__ kmat,
                               const int* __restrict__ src, const int* __restrict__ dst,
                               float* __restrict__ sbuf, unsigned* __restrict__ mkey, int E) {
    int wid  = blockIdx.x * (blockDim.x >> 5) + (threadIdx.x >> 5);
    int lane = threadIdx.x & 31;
    if (wid >= E) return;
    int s = src[wid], d = dst[wid];
    const float4* qp = (const float4*)(qW   + (size_t)s * DD);
    const float4* kp = (const float4*)(kmat + (size_t)d * DD);
    float4 qv = qp[lane], kv = kp[lane];
    float p = qv.x * kv.x + qv.y * kv.y + qv.z * kv.z + qv.w * kv.w;
#pragma unroll
    for (int off = 16; off > 0; off >>= 1) p += __shfl_xor(p, off, 32);
    if (lane == 0) {
        float sc = p * 0.08838834764831845f;           // 1/sqrt(128)
        sc = sc > 0.f ? sc : 0.01f * sc;               // leaky_relu(0.01)
        sbuf[wid] = sc;
        atomicMax(&mkey[d], f2ord(sc));
    }
}

// Thread per edge: e = exp(s - m[dst]); den[dst] += e
__global__ void gnn_edge_exp(const float* __restrict__ sbuf, const unsigned* __restrict__ mkey,
                             const int* __restrict__ dst, float* __restrict__ ebuf,
                             float* __restrict__ den, int E) {
    int e = blockIdx.x * blockDim.x + threadIdx.x;
    if (e >= E) return;
    int d = dst[e];
    float w = __expf(sbuf[e] - ord2f(mkey[d]));
    ebuf[e] = w;
    atomicAdd(&den[d], w);
}

// Wave per edge: out[dst] += (e/den[dst]) * v[src]
__global__ void gnn_edge_scatter(const float* __restrict__ ebuf, const float* __restrict__ den,
                                 const float* __restrict__ v, const int* __restrict__ src,
                                 const int* __restrict__ dst, float* __restrict__ out, int E) {
    int wid  = blockIdx.x * (blockDim.x >> 5) + (threadIdx.x >> 5);
    int lane = threadIdx.x & 31;
    if (wid >= E) return;
    int s = src[wid], d = dst[wid];
    float alpha = ebuf[wid] / den[d];
    const float4* vp = (const float4*)(v + (size_t)s * DD);
    float4 vv = vp[lane];
    float* op = out + (size_t)d * DD + lane * 4;
    atomicAdd(op + 0, alpha * vv.x);
    atomicAdd(op + 1, alpha * vv.y);
    atomicAdd(op + 2, alpha * vv.z);
    atomicAdd(op + 3, alpha * vv.w);
}

// ---------------------------------------------------------------------------
// Utility kernels
// ---------------------------------------------------------------------------
__global__ void gnn_fill_f32(float* p, float v, size_t n) {
    size_t i = (size_t)blockIdx.x * blockDim.x + threadIdx.x;
    if (i < n) p[i] = v;
}
__global__ void gnn_fill_u32(unsigned* p, unsigned v, size_t n) {
    size_t i = (size_t)blockIdx.x * blockDim.x + threadIdx.x;
    if (i < n) p[i] = v;
}
__global__ void gnn_relu(float* p, size_t n) {
    size_t i = (size_t)blockIdx.x * blockDim.x + threadIdx.x;
    if (i < n) p[i] = fmaxf(p[i], 0.f);
}

// ---------------------------------------------------------------------------
// Launcher
// ---------------------------------------------------------------------------
extern "C" void kernel_launch(void* const* d_in, const int* in_sizes, int n_in,
                              void* d_out, int out_size, void* d_ws, size_t ws_size,
                              hipStream_t stream) {
    const float* x_a    = (const float*)d_in[0];
    const float* x_b    = (const float*)d_in[1];
    const int*   ei_ab  = (const int*)d_in[2];
    const int*   ei_ba  = (const int*)d_in[3];
    const float* qkv_w  = (const float*)d_in[4];   // [2,2,3,D,D]
    const float* qkv_b  = (const float*)d_in[5];   // [2,2,3,D]
    const float* edge_W = (const float*)d_in[6];   // [2,2,D,D]

    const int N = in_sizes[0] / DD;
    const int E = in_sizes[2] / 2;
    const size_t ND = (size_t)N * DD;

    // Workspace layout (all 4-byte elements)
    float* ws   = (float*)d_ws;
    float* qW_a = ws + 0 * ND;
    float* k_a  = ws + 1 * ND;
    float* v_a  = ws + 2 * ND;
    float* qW_b = ws + 3 * ND;
    float* k_b  = ws + 4 * ND;
    float* v_b  = ws + 5 * ND;
    float* h_a  = ws + 6 * ND;
    float* h_b  = ws + 7 * ND;
    float* sbuf = ws + 8 * ND;
    float* ebuf = sbuf + E;
    unsigned* mkey = (unsigned*)(ebuf + E);
    float* den  = (float*)(mkey + N);
    float* Wf_a = den + N;
    float* Wf_b = Wf_a + DD * DD;
    float* bf_a = Wf_b + DD * DD;
    float* bf_b = bf_a + DD;

    const int gemmBlocksN = (N + 63) / 64;
    const int fillN_ND    = (int)((ND + 255) / 256);
    const int fillN_N     = (N + 255) / 256;
    const int edgeWaveBlk = (E + 7) / 8;       // 256 threads = 8 waves/block
    const int edgeThrBlk  = (E + 255) / 256;

    auto edge_attend = [&](const int* ei, const float* qW, const float* kmat,
                           const float* vmat, float* out) {
        gnn_fill_u32<<<fillN_N, 256, 0, stream>>>(mkey, 0u, (size_t)N);
        gnn_fill_f32<<<fillN_N, 256, 0, stream>>>(den, 0.f, (size_t)N);
        gnn_edge_score<<<edgeWaveBlk, 256, 0, stream>>>(qW, kmat, ei, ei + E, sbuf, mkey, E);
        gnn_edge_exp<<<edgeThrBlk, 256, 0, stream>>>(sbuf, mkey, ei + E, ebuf, den, E);
        gnn_edge_scatter<<<edgeWaveBlk, 256, 0, stream>>>(ebuf, den, vmat, ei, ei + E, out, E);
    };

    auto run_layer = [&](int l, const float* xa, const float* xb,
                         float* out_a, float* out_b) {
        const size_t wDD = (size_t)DD * DD;
        const float* Wq_a = qkv_w + (((size_t)l * 2 + 0) * 3 + 0) * wDD;
        const float* Wk_a = qkv_w + (((size_t)l * 2 + 0) * 3 + 1) * wDD;
        const float* Wv_a = qkv_w + (((size_t)l * 2 + 0) * 3 + 2) * wDD;
        const float* Wq_b = qkv_w + (((size_t)l * 2 + 1) * 3 + 0) * wDD;
        const float* Wk_b = qkv_w + (((size_t)l * 2 + 1) * 3 + 1) * wDD;
        const float* Wv_b = qkv_w + (((size_t)l * 2 + 1) * 3 + 2) * wDD;
        const float* bq_a = qkv_b + (((size_t)l * 2 + 0) * 3 + 0) * DD;
        const float* bk_a = qkv_b + (((size_t)l * 2 + 0) * 3 + 1) * DD;
        const float* bv_a = qkv_b + (((size_t)l * 2 + 0) * 3 + 2) * DD;
        const float* bq_b = qkv_b + (((size_t)l * 2 + 1) * 3 + 0) * DD;
        const float* bk_b = qkv_b + (((size_t)l * 2 + 1) * 3 + 1) * DD;
        const float* bv_b = qkv_b + (((size_t)l * 2 + 1) * 3 + 2) * DD;
        const float* eW_ab = edge_W + ((size_t)l * 2 + 0) * wDD;
        const float* eW_ba = edge_W + ((size_t)l * 2 + 1) * wDD;

        // Fold the bilinear weight into the q projection: Wf = Wq @ eW, bf = bq @ eW
        gnn_gemm3<<<2, 128, 0, stream>>>(Wq_a, eW_ab, nullptr, nullptr,
                                         nullptr, nullptr, nullptr,
                                         Wf_a, nullptr, nullptr, DD);
        gnn_gemm3<<<2, 128, 0, stream>>>(Wq_b, eW_ba, nullptr, nullptr,
                                         nullptr, nullptr, nullptr,
                                         Wf_b, nullptr, nullptr, DD);
        gnn_bias_fuse<<<1, 128, 0, stream>>>(bq_a, eW_ab, bf_a);
        gnn_bias_fuse<<<1, 128, 0, stream>>>(bq_b, eW_ba, bf_b);

        // Fused node projections: qW, k, v in one pass over x (bf16 WMMA)
        gnn_gemm3<<<gemmBlocksN, 128, 0, stream>>>(xa, Wf_a, Wk_a, Wv_a,
                                                   bf_a, bk_a, bv_a,
                                                   qW_a, k_a, v_a, N);
        gnn_gemm3<<<gemmBlocksN, 128, 0, stream>>>(xb, Wf_b, Wk_b, Wv_b,
                                                   bf_b, bk_b, bv_b,
                                                   qW_b, k_b, v_b, N);

        gnn_fill_f32<<<fillN_ND, 256, 0, stream>>>(out_a, 0.f, ND);
        gnn_fill_f32<<<fillN_ND, 256, 0, stream>>>(out_b, 0.f, ND);

        // a->b edges: q/v from src 'a', k from dst 'b', accumulate into out_b
        edge_attend(ei_ab, qW_a, k_b, v_a, out_b);
        // b->a edges: q/v from src 'b', k from dst 'a', accumulate into out_a
        edge_attend(ei_ba, qW_b, k_a, v_b, out_a);
    };

    float* o_a = (float*)d_out;
    float* o_b = o_a + ND;

    run_layer(0, x_a, x_b, h_a, h_b);
    gnn_relu<<<fillN_ND, 256, 0, stream>>>(h_a, ND);
    gnn_relu<<<fillN_ND, 256, 0, stream>>>(h_b, ND);
    run_layer(1, h_a, h_b, o_a, o_b);
}
